// Edge_Model_23562190586356
// MI455X (gfx1250) — matrix operand order repaired and checked by reading, not compiled, not run on hardware
//
#include <hip/hip_runtime.h>

// ---------------- problem constants (from reference) ----------------
#define E_TOTAL 1600000
#define NBATCH  1024
#define FX      64
#define FE      32
#define FU      32
#define HID     128
#define KIN     160      // src(64) + dst(64) + edge_attr(32); u-term folded into ug
#define OUTC    32
#define TILE_M  128      // edges per workgroup
#define NTHREADS 256     // 8 wave32s

typedef __attribute__((ext_vector_type(16))) __bf16 v16bf;
typedef __attribute__((ext_vector_type(8)))  float  v8f;
// native vector type matching the async-to-LDS builtin's signature
typedef __attribute__((ext_vector_type(4)))  int    i4;
typedef __attribute__((address_space(1)))    i4     gi4;   // global
typedef __attribute__((address_space(3)))    i4     li4;   // LDS

// exact round-to-nearest-even f32 -> bf16 (used in one-time weight prep)
__device__ __forceinline__ unsigned int f2bf(float f) {
    unsigned int u = __float_as_uint(f);
    return (u + 0x7FFFu + ((u >> 16) & 1u)) >> 16;
}
// fast pack: two f32 -> one dword of bf16x2, round-half-up then v_perm_b32
// (3 VALU total; differs from RNE only on exact ties)
__device__ __forceinline__ unsigned int f2bf_pk(float lo, float hi) {
    unsigned int ulo = __float_as_uint(lo) + 0x8000u;
    unsigned int uhi = __float_as_uint(hi) + 0x8000u;
    return __builtin_amdgcn_perm(uhi, ulo, 0x07060302u);  // [uhi.hi16 | ulo.hi16]
}
// single-instruction ReLU: v_med3_num_f32(x, 0, +inf)
__device__ __forceinline__ float relu(float x) {
    return __builtin_amdgcn_fmed3f(x, 0.0f, __builtin_inff());
}

// ---------------- prep: ug[b][j] = b1[j] + sum_k u[b][k] * W_u[k][j] ----------------
__global__ void prep_ug_kernel(const float* __restrict__ u,
                               const float* __restrict__ W1,
                               const float* __restrict__ b1,
                               float* __restrict__ ug) {
    const int b = blockIdx.x;       // 1024 blocks
    const int j = threadIdx.x;      // 128 threads
    float acc = b1[j];
    #pragma unroll 8
    for (int k = 0; k < FU; ++k)
        acc += u[b * FU + k] * W1[(2 * FX + FE + k) * HID + j];
    ug[b * HID + j] = acc;
}

// ---------------- prep: transpose + convert weights to bf16 (exact RNE) ----------------
// W1t[n][k] = bf16(W1[k][n]), n<128, k<160   (N-major; used as WMMA A fragments)
// W2t[n][k] = bf16(W2[k][n]), n<32,  k<128
__global__ void prep_w_kernel(const float* __restrict__ W1,
                              const float* __restrict__ W2,
                              unsigned short* __restrict__ W1t,
                              unsigned short* __restrict__ W2t) {
    const int i = blockIdx.x * NTHREADS + threadIdx.x;
    if (i < HID * KIN) {
        const int n = i / KIN, k = i % KIN;
        W1t[i] = (unsigned short)f2bf(W1[k * HID + n]);
    }
    if (i < OUTC * HID) {
        const int n = i / HID, k = i % HID;
        W2t[i] = (unsigned short)f2bf(W2[k * OUTC + n]);
    }
}

// ---------------- main fused edge-MLP kernel ----------------
// GEMMs are computed TRANSPOSED: D[chan][edge] = W[chan][k] (A) x Act^T[k][edge] (B).
// This makes every lane's 8 D-elements 8 *contiguous channels of one edge*, so all
// epilogues / C-inits are contiguous vector accesses.
//
// LDS layout (dynamic shared):
//   sA  : [128][160] bf16  (packed [src|dst|e] rows)        40960 B @ 0
//         (reused after GEMM1 as sOut: [128][32] f32,       16384 B @ 0)
//   sW1 : [128][160] bf16  (W1t, N-major)                   40960 B @ 40960
//   sH  : [128][128] bf16  (relu hidden, edge-major)        32768 B @ 81920
//   sW2 : [ 32][128] bf16  (W2t, N-major)                    8192 B @ 114688
//   sB  : [128] int        (batch ids of this tile)           512 B @ 122880
#define SMEM_BYTES 123392

__global__ void __launch_bounds__(NTHREADS)
edge_mlp_kernel(const float* __restrict__ src,
                const float* __restrict__ dst,
                const float* __restrict__ eattr,
                const int*   __restrict__ batch,
                const float* __restrict__ ug,     // [1024][128] f32, b1 folded in
                const unsigned short* __restrict__ W1t,
                const unsigned short* __restrict__ W2t,
                const float* __restrict__ b2,
                float* __restrict__ out) {
    extern __shared__ char smem[];
    unsigned short* sA   = (unsigned short*)(smem);
    float*          sOut = (float*)(smem);            // aliases sA, used after GEMM1
    unsigned short* sW1  = (unsigned short*)(smem + 40960);
    unsigned short* sH   = (unsigned short*)(smem + 81920);
    unsigned short* sW2  = (unsigned short*)(smem + 114688);
    int*            sB   = (int*)(smem + 122880);

    const int tid  = threadIdx.x;
    const int lane = tid & 31;
    const int wave = tid >> 5;          // 0..7, each wave owns 16 edges (B columns)
    const int half = lane >> 4;         // half-wave select (WMMA fragment layout)
    const int l16  = lane & 15;
    const long long ebase = (long long)blockIdx.x * TILE_M;

    // ---- kick off async weight staging first (overlaps with bf16 conversion below) ----
#if __has_builtin(__builtin_amdgcn_global_load_async_to_lds_b128)
    {
        i4* gw1 = (i4*)(void*)W1t;
        i4* gw2 = (i4*)(void*)W2t;
        i4* lw1 = (i4*)sW1;
        i4* lw2 = (i4*)sW2;
        for (int i = tid; i < (HID * KIN) / 8; i += NTHREADS)
            __builtin_amdgcn_global_load_async_to_lds_b128(
                (gi4*)(gw1 + i), (li4*)(lw1 + i), 0, 0);
        for (int i = tid; i < (OUTC * HID) / 8; i += NTHREADS)
            __builtin_amdgcn_global_load_async_to_lds_b128(
                (gi4*)(gw2 + i), (li4*)(lw2 + i), 0, 0);
    }
#else
    for (int i = tid; i < (HID * KIN) / 8; i += NTHREADS)
        ((uint4*)sW1)[i] = ((const uint4*)W1t)[i];
    for (int i = tid; i < (OUTC * HID) / 8; i += NTHREADS)
        ((uint4*)sW2)[i] = ((const uint4*)W2t)[i];
#endif

    // ---- stage edge features: pack [src|dst|e] rows as bf16 (float4 granularity) ----
    // 128 rows x 40 float4 slots: slot<16 -> src, <32 -> dst, else eattr
    for (int i = tid; i < TILE_M * 40; i += NTHREADS) {
        const int row = i / 40, s = i % 40;
        const long long e = ebase + row;
        float4 v; int kbase;
        if (s < 16)      { v = ((const float4*)(src   + e * FX))[s];      kbase = s * 4; }
        else if (s < 32) { v = ((const float4*)(dst   + e * FX))[s - 16]; kbase = FX + (s - 16) * 4; }
        else             { v = ((const float4*)(eattr + e * FE))[s - 32]; kbase = 2 * FX + (s - 32) * 4; }
        uint2 pk;
        pk.x = f2bf_pk(v.x, v.y);
        pk.y = f2bf_pk(v.z, v.w);
        *(uint2*)(sA + row * KIN + kbase) = pk;
    }
    if (tid < TILE_M) sB[tid] = batch[ebase + tid];

#if __has_builtin(__builtin_amdgcn_global_load_async_to_lds_b128)
  #if __has_builtin(__builtin_amdgcn_s_wait_asynccnt)
    __builtin_amdgcn_s_wait_asynccnt(0);
  #else
    asm volatile("s_wait_asynccnt 0x0" ::: "memory");
  #endif
#endif
    __syncthreads();

    // this lane's edge (B column index == D column index), constant for whole kernel
    const int edge16 = wave * 16 + l16;
    const float* ugp = ug + (long long)sB[edge16] * HID + 8 * half;

    // ================= GEMM1: h^T tile = W1t (A) x sA^T (B) + ug[batch] =================
    // A bf16 16x32 fragment: row = ncol + l16 (hidden chan),
    //   16 contiguous K at k0 + 16*half
    // B bf16 32x16 fragment: col = edge16, 16 contiguous K at k0 + 16*half
    // C/D: element v -> (chan = ncol + v + 8*half, edge = edge16)
    #pragma unroll
    for (int nt = 0; nt < 8; ++nt) {
        const int ncol = nt * 16;
        v8f c;
        {   // C-init: 8 contiguous floats of one ug row -> two b128 loads
            const float4 c0 = *(const float4*)(ugp + ncol);
            const float4 c1 = *(const float4*)(ugp + ncol + 4);
            c[0] = c0.x; c[1] = c0.y; c[2] = c0.z; c[3] = c0.w;
            c[4] = c1.x; c[5] = c1.y; c[6] = c1.z; c[7] = c1.w;
        }
        #pragma unroll
        for (int kc = 0; kc < KIN / 32; ++kc) {   // 5 WMMAs
            const int k0 = kc * 32;
            union { v16bf v; uint4 q[2]; } a, b;
            const unsigned short* pa = sW1 + (ncol + l16) * KIN + k0 + half * 16;
            a.q[0] = *(const uint4*)(pa);
            a.q[1] = *(const uint4*)(pa + 8);
            const unsigned short* pb = sA + edge16 * KIN + k0 + half * 16;
            b.q[0] = *(const uint4*)(pb);
            b.q[1] = *(const uint4*)(pb + 8);
            c = __builtin_amdgcn_wmma_f32_16x16x32_bf16(
                    false, a.v, false, b.v, (short)0, c, false, false);
        }
        // relu (v_med3) + pack (v_perm) 8 contiguous channels -> single ds_store_b128
        uint4 pk;
        pk.x = f2bf_pk(relu(c[0]), relu(c[1]));
        pk.y = f2bf_pk(relu(c[2]), relu(c[3]));
        pk.z = f2bf_pk(relu(c[4]), relu(c[5]));
        pk.w = f2bf_pk(relu(c[6]), relu(c[7]));
        *(uint4*)(sH + edge16 * HID + ncol + 8 * half) = pk;
    }
    __syncthreads();   // sH complete; sA region now dead -> reuse as sOut

    // ================= GEMM2: out^T tile = W2t (A) x sH^T (B) + b2 =================
    #pragma unroll
    for (int nt = 0; nt < 2; ++nt) {
        const int ncol = nt * 16;
        v8f c;
        {   // bias: 8 contiguous output channels
            const float4 b0 = *(const float4*)(b2 + ncol + 8 * half);
            const float4 b1v = *(const float4*)(b2 + ncol + 8 * half + 4);
            c[0] = b0.x; c[1] = b0.y; c[2] = b0.z; c[3] = b0.w;
            c[4] = b1v.x; c[5] = b1v.y; c[6] = b1v.z; c[7] = b1v.w;
        }
        #pragma unroll
        for (int kc = 0; kc < HID / 32; ++kc) {   // 4 WMMAs
            const int k0 = kc * 32;
            union { v16bf v; uint4 q[2]; } a, b;
            const unsigned short* pa = sW2 + (ncol + l16) * HID + k0 + half * 16;
            a.q[0] = *(const uint4*)(pa);
            a.q[1] = *(const uint4*)(pa + 8);
            const unsigned short* pb = sH + edge16 * HID + k0 + half * 16;
            b.q[0] = *(const uint4*)(pb);
            b.q[1] = *(const uint4*)(pb + 8);
            c = __builtin_amdgcn_wmma_f32_16x16x32_bf16(
                    false, a.v, false, b.v, (short)0, c, false, false);
        }
        // 8 contiguous output channels of one edge -> two b128 LDS stores
        float* po = sOut + edge16 * OUTC + ncol + 8 * half;
        *(float4*)(po)     = make_float4(c[0], c[1], c[2], c[3]);
        *(float4*)(po + 4) = make_float4(c[4], c[5], c[6], c[7]);
    }
    __syncthreads();

    // ---- coalesced write-back: 128 edges x 32 f32 = 16 KB contiguous ----
    {
        uint4*       dstp = (uint4*)(out + ebase * OUTC);
        const uint4* srcp = (const uint4*)sOut;
        #pragma unroll
        for (int i = tid; i < (TILE_M * OUTC) / 4; i += NTHREADS)
            dstp[i] = srcp[i];
    }
}

// ---------------- launch ----------------
extern "C" void kernel_launch(void* const* d_in, const int* in_sizes, int n_in,
                              void* d_out, int out_size, void* d_ws, size_t ws_size,
                              hipStream_t stream) {
    const float* src   = (const float*)d_in[0];
    const float* dst   = (const float*)d_in[1];
    const float* eattr = (const float*)d_in[2];
    const float* u     = (const float*)d_in[3];
    const int*   batch = (const int*)  d_in[4];
    const float* W1    = (const float*)d_in[5];
    const float* b1    = (const float*)d_in[6];
    const float* W2    = (const float*)d_in[7];
    const float* b2    = (const float*)d_in[8];
    float* out = (float*)d_out;

    char* ws = (char*)d_ws;
    float*          ug  = (float*)ws;                             // 1024*128*4 = 524288 B
    unsigned short* W1t = (unsigned short*)(ws + 524288);         // 128*160*2  =  40960 B
    unsigned short* W2t = (unsigned short*)(ws + 524288 + 40960); // 32*128*2   =   8192 B

    (void)hipFuncSetAttribute((const void*)edge_mlp_kernel,
                              hipFuncAttributeMaxDynamicSharedMemorySize, SMEM_BYTES);

    prep_ug_kernel<<<NBATCH, HID, 0, stream>>>(u, W1, b1, ug);
    prep_w_kernel<<<(HID * KIN + NTHREADS - 1) / NTHREADS, NTHREADS, 0, stream>>>(W1, W2, W1t, W2t);
    edge_mlp_kernel<<<E_TOTAL / TILE_M, NTHREADS, SMEM_BYTES, stream>>>(
        src, dst, eattr, batch, ug, W1t, W2t, b2, out);
}
